// SparseAttention_50964081934689
// MI455X (gfx1250) — matrix operand order repaired and checked by reading, not compiled, not run on hardware
//
#include <hip/hip_runtime.h>
#include <hip/hip_bf16.h>

typedef __bf16 bf16_t;
typedef __bf16  v16bf __attribute__((ext_vector_type(16)));
typedef __bf16  v8bf  __attribute__((ext_vector_type(8)));
typedef float   v8f   __attribute__((ext_vector_type(8)));
typedef unsigned int u32x4 __attribute__((ext_vector_type(4)));
typedef int          i32x4 __attribute__((ext_vector_type(4)));
typedef int          i32x8 __attribute__((ext_vector_type(8)));

// ---------------------------------------------------------------------------
// fp32 -> bf16 conversion
// ---------------------------------------------------------------------------
__global__ __launch_bounds__(256)
void cvt_f32_bf16_kernel(const float* __restrict__ in, bf16_t* __restrict__ out,
                         long long n) {
    long long i = (long long)blockIdx.x * blockDim.x + threadIdx.x;
    if (i < n) out[i] = (bf16_t)in[i];
}

// ---------------------------------------------------------------------------
// Fragment loaders for v_wmma_f32_16x16x32_bf16 (NT layout, K contiguous)
// A frag: lane L holds row M=L%16; elems 0-15 <- K=(L/16)*16..+15
// B frag: lane L holds col N=L%16; elems 0-15 <- K=(L/16)*16..+15
// ---------------------------------------------------------------------------
__device__ __forceinline__ v16bf load_a_frag(const bf16_t* p) {
    v8bf lo = *(const v8bf*)(p);
    v8bf hi = *(const v8bf*)(p + 8);
    return __builtin_shufflevector(lo, hi, 0,1,2,3,4,5,6,7,8,9,10,11,12,13,14,15);
}
__device__ __forceinline__ v16bf load_b_frag(const bf16_t* p) {
    return *(const v16bf*)(p);
}
__device__ __forceinline__ v8f wmma_bf16(v16bf a, v16bf b, v8f c) {
    return __builtin_amdgcn_wmma_f32_16x16x32_bf16(false, a, false, b, (short)0, c,
                                                   false, false);
}

// ---------------------------------------------------------------------------
// TDM: DMA a 2-D tile (tile_dim0=32 elems along K contiguous, tile_dim1=128
// rows) of a bf16 tensor (row length tensor_d0, row stride d0_stride elems,
// tensor_d1 rows) from global memory into LDS at byte offset lds_off.
// D# packing per CDNA5 ISA ch. 8 (group0: count/lds/global/type,
// group1: data_size=2B, dims, tile dims, strides; groups 2/3 zero => 2-D).
// ---------------------------------------------------------------------------
__device__ __forceinline__ void tdm_load_tile_2d(unsigned lds_off,
                                                 const bf16_t* gptr,
                                                 unsigned tensor_d0,
                                                 unsigned tensor_d1,
                                                 unsigned long long d0_stride) {
    unsigned long long ga = (unsigned long long)(size_t)gptr;
    u32x4 g0 = { 1u,                                      // count=1 (valid)
                 lds_off,                                 // lds_addr (bytes)
                 (unsigned)(ga & 0xffffffffu),            // global_addr[31:0]
                 (unsigned)((ga >> 32) & 0x1ffffffu) | (2u << 30) }; // [56:32]|type=2
    i32x8 g1 = {
        (int)(1u << 16),                                            // data_size=2B
        (int)((tensor_d0 & 0xffffu) << 16),                         // dim0[15:0]
        (int)(((tensor_d0 >> 16) & 0xffffu) |
              ((tensor_d1 & 0xffffu) << 16)),                       // dim0[31:16]|dim1[15:0]
        (int)(((tensor_d1 >> 16) & 0xffffu) | (32u << 16)),         // dim1[31:16]|tile_dim0=32
        (int)(128u),                                                // tile_dim1=128, tile_dim2=0
        (int)(unsigned)(d0_stride & 0xffffffffu),                   // dim0_stride[31:0]
        (int)(unsigned)((d0_stride >> 32) & 0xffffu),               // dim0_stride[47:32]
        0 };
    i32x4 z4 = {0, 0, 0, 0};
#if __clang_major__ >= 23
    i32x8 z8 = {0, 0, 0, 0, 0, 0, 0, 0};
    __builtin_amdgcn_tensor_load_to_lds(g0, g1, z4, z4, z8, 0);
#else
    __builtin_amdgcn_tensor_load_to_lds(g0, g1, z4, z4, 0);
#endif
    asm volatile("" ::: "memory");
}

// ---------------------------------------------------------------------------
// NT GEMM with TDM-staged LDS tiles: C[m,n] = sum_k A[m,k]*B[n,k] + bias[n]
// A: M x K bf16 row-major, B: N x K bf16 row-major (computes A @ B^T).
// Block tile 128x128, 8 waves (4 along M x 2 along N), wave tile 32x64.
// Per K-step (32): TDM loads A-tile (128x32) and B-tile (128x32) into LDS,
// double-buffered; waves read ds fragments and issue 8 WMMAs each.
// OUT_MODE 0: bf16 [m,n] | 1: bf16 [m,(n%128)*32+n/128] (V) | 2: f32 [m,n]
// ---------------------------------------------------------------------------
template <int OUT_MODE>
__global__ __launch_bounds__(256)
void gemm_nt_wmma(const bf16_t* __restrict__ A, const bf16_t* __restrict__ B,
                  const float* __restrict__ bias, void* __restrict__ Cout,
                  int M, int N, int K) {
    __shared__ bf16_t ldsA[2][128 * 32];   // 2 x 8KB
    __shared__ bf16_t ldsB[2][128 * 32];   // 2 x 8KB

    const int lane = threadIdx.x & 31;
    const int wave = threadIdx.x >> 5;      // 0..7
    const int wm   = wave & 3;              // 4 waves along M
    const int wn   = wave >> 2;             // 2 waves along N
    const int l16  = lane & 15;
    const int ka   = (lane >> 4) << 4;      // A-frag K sub-offset (0 or 16)
    const int kbo  = (lane >> 4) << 4;      // B-frag K sub-offset (0 or 16)
    const int rofs = (lane >> 4) << 3;      // C row sub-offset (0 or 8)

    const int mblk = blockIdx.y * 128;
    const int nblk = blockIdx.x * 128;
    const int m0   = mblk + wm * 32;        // 2 M-subtiles of 16
    const int n0   = nblk + wn * 64;        // 4 N-subtiles of 16
    const int steps = K >> 5;               // K/32

    const unsigned ldsA0 = (unsigned)(size_t)&ldsA[0][0];
    const unsigned ldsA1 = (unsigned)(size_t)&ldsA[1][0];
    const unsigned ldsB0 = (unsigned)(size_t)&ldsB[0][0];
    const unsigned ldsB1 = (unsigned)(size_t)&ldsB[1][0];

    v8f acc[2][4];
    const v8f vz = {0.f, 0.f, 0.f, 0.f, 0.f, 0.f, 0.f, 0.f};
#pragma unroll
    for (int mi = 0; mi < 2; ++mi)
#pragma unroll
        for (int ni = 0; ni < 4; ++ni) acc[mi][ni] = vz;

    // ---- prologue: post tiles for steps 0 and 1, wait for step 0 ----
    if (wave == 0) {
        tdm_load_tile_2d(ldsA0, A + (size_t)mblk * K + 0,  K, M, (unsigned long long)K);
        tdm_load_tile_2d(ldsB0, B + (size_t)nblk * K + 0,  K, N, (unsigned long long)K);
        if (steps > 1) {
            tdm_load_tile_2d(ldsA1, A + (size_t)mblk * K + 32, K, M, (unsigned long long)K);
            tdm_load_tile_2d(ldsB1, B + (size_t)nblk * K + 32, K, N, (unsigned long long)K);
            __builtin_amdgcn_s_wait_tensorcnt((short)2);
        } else {
            __builtin_amdgcn_s_wait_tensorcnt((short)0);
        }
        asm volatile("" ::: "memory");
    }
    __syncthreads();

#pragma unroll 1
    for (int i = 0; i < steps; ++i) {
        const int buf = i & 1;
        const bf16_t* aT = &ldsA[buf][0];
        const bf16_t* bT = &ldsB[buf][0];

        v16bf af[2], bfr[4];
#pragma unroll
        for (int mi = 0; mi < 2; ++mi)
            af[mi] = load_a_frag(aT + (wm * 32 + mi * 16 + l16) * 32 + ka);
#pragma unroll
        for (int ni = 0; ni < 4; ++ni)
            bfr[ni] = load_b_frag(bT + (wn * 64 + ni * 16 + l16) * 32 + kbo);
#pragma unroll
        for (int mi = 0; mi < 2; ++mi)
#pragma unroll
            for (int ni = 0; ni < 4; ++ni)
                acc[mi][ni] = wmma_bf16(af[mi], bfr[ni], acc[mi][ni]);

        __syncthreads();   // all waves done reading buf before TDM overwrites it
        if (wave == 0) {
            if (i + 2 < steps) {
                const unsigned la = buf ? ldsA1 : ldsA0;
                const unsigned lb = buf ? ldsB1 : ldsB0;
                const int koff = (i + 2) * 32;
                tdm_load_tile_2d(la, A + (size_t)mblk * K + koff, K, M, (unsigned long long)K);
                tdm_load_tile_2d(lb, B + (size_t)nblk * K + koff, K, N, (unsigned long long)K);
                __builtin_amdgcn_s_wait_tensorcnt((short)2);  // next buffer ready
            } else if (i + 1 < steps) {
                __builtin_amdgcn_s_wait_tensorcnt((short)0);
            }
            asm volatile("" ::: "memory");
        }
        __syncthreads();   // release waves to read the next buffer
    }

    // ---- epilogue: bias + store ----
#pragma unroll
    for (int ni = 0; ni < 4; ++ni) {
        const int n = n0 + ni * 16 + l16;
        const float bv = bias ? bias[n] : 0.f;
#pragma unroll
        for (int mi = 0; mi < 2; ++mi) {
            v8f c = acc[mi][ni];
#pragma unroll
            for (int r = 0; r < 8; ++r) {
                const int m = m0 + mi * 16 + rofs + r;
                const float val = c[r] + bv;
                if (OUT_MODE == 0) {
                    ((bf16_t*)Cout)[(size_t)m * N + n] = (bf16_t)val;
                } else if (OUT_MODE == 1) {
                    ((bf16_t*)Cout)[(size_t)m * N + (n & 127) * 32 + (n >> 7)] =
                        (bf16_t)val;
                } else {
                    ((float*)Cout)[(size_t)m * N + n] = val;
                }
            }
        }
    }
}

// ---------------------------------------------------------------------------
// Per-token across-head attention. One wave per token, 4 waves per block.
// Q, Kmat: [token][h*128 + d] bf16 ; Vt: [token][d*32 + g] bf16
// scores = (Q_t @ K_t^T)/sqrt(128); mask s>1e-3 else -inf; softmax over g;
// out = attn @ V_t
// ---------------------------------------------------------------------------
__global__ __launch_bounds__(128)
void attention_wmma_kernel(const bf16_t* __restrict__ Q,
                           const bf16_t* __restrict__ Kmat,
                           const bf16_t* __restrict__ Vt,
                           bf16_t* __restrict__ Out) {
    __shared__ float  sc[4][32][33];   // padded f32 scores
    __shared__ bf16_t aw[4][32][32];   // bf16 attn weights (64B rows)

    const int wave = threadIdx.x >> 5;
    const int lane = threadIdx.x & 31;
    const int l16  = lane & 15;
    const int ka   = (lane >> 4) << 4;
    const int kbo  = (lane >> 4) << 4;
    const int rofs = (lane >> 4) << 3;
    const size_t token = (size_t)blockIdx.x * 4 + wave;

    const bf16_t* q = Q    + token * 4096;
    const bf16_t* k = Kmat + token * 4096;
    const bf16_t* v = Vt   + token * 4096;

    const v8f vz = {0.f, 0.f, 0.f, 0.f, 0.f, 0.f, 0.f, 0.f};
    v8f acc[2][2];
#pragma unroll
    for (int mi = 0; mi < 2; ++mi)
#pragma unroll
        for (int gi = 0; gi < 2; ++gi) acc[mi][gi] = vz;

#pragma unroll
    for (int kk = 0; kk < 128; kk += 32) {
        v16bf qa[2], kb[2];
#pragma unroll
        for (int mi = 0; mi < 2; ++mi)
            qa[mi] = load_a_frag(q + (mi * 16 + l16) * 128 + kk + ka);
#pragma unroll
        for (int gi = 0; gi < 2; ++gi)
            kb[gi] = load_b_frag(k + (gi * 16 + l16) * 128 + kk + kbo);
#pragma unroll
        for (int mi = 0; mi < 2; ++mi)
#pragma unroll
            for (int gi = 0; gi < 2; ++gi)
                acc[mi][gi] = wmma_bf16(qa[mi], kb[gi], acc[mi][gi]);
    }

    const float scale = 0.08838834764831845f;  // 1/sqrt(128)
#pragma unroll
    for (int mi = 0; mi < 2; ++mi)
#pragma unroll
        for (int gi = 0; gi < 2; ++gi)
#pragma unroll
            for (int r = 0; r < 8; ++r)
                sc[wave][mi * 16 + rofs + r][gi * 16 + l16] =
                    acc[mi][gi][r] * scale;

    __syncthreads();

    // masked softmax over g; lane handles row h = lane
    {
        const int h = lane;
        float mx = -1.0e30f;
#pragma unroll
        for (int g = 0; g < 32; ++g) {
            float s = sc[wave][h][g];
            if (s > 0.001f && s > mx) mx = s;
        }
        float p[32];
        float sum = 0.f;
#pragma unroll
        for (int g = 0; g < 32; ++g) {
            float s = sc[wave][h][g];
            float e = (s > 0.001f) ? __expf(s - mx) : 0.f;
            p[g] = e;
            sum += e;
        }
        const float inv = 1.f / sum;
#pragma unroll
        for (int g = 0; g < 32; ++g)
            aw[wave][h][g] = (bf16_t)(p[g] * inv);
    }

    __syncthreads();

    // out = attn(32x32) @ V(32x128): K=32 single WMMA, 2x8 tiles
    v16bf a2[2];
#pragma unroll
    for (int mi = 0; mi < 2; ++mi)
        a2[mi] = load_a_frag(&aw[wave][mi * 16 + l16][ka]);

#pragma unroll
    for (int ni = 0; ni < 8; ++ni) {
        const v16bf bfr = load_b_frag(v + (ni * 16 + l16) * 32 + kbo);
#pragma unroll
        for (int mi = 0; mi < 2; ++mi) {
            v8f o = wmma_bf16(a2[mi], bfr, vz);
#pragma unroll
            for (int r = 0; r < 8; ++r) {
                const int h = mi * 16 + rofs + r;
                const int d = ni * 16 + l16;
                Out[token * 4096 + h * 128 + d] = (bf16_t)o[r];
            }
        }
    }
}

// ---------------------------------------------------------------------------
// Host launcher
// ---------------------------------------------------------------------------
extern "C" void kernel_launch(void* const* d_in, const int* in_sizes, int n_in,
                              void* d_out, int out_size, void* d_ws, size_t ws_size,
                              hipStream_t stream) {
    (void)in_sizes; (void)n_in; (void)out_size; (void)ws_size;
    const float* x  = (const float*)d_in[0];
    const float* Wq = (const float*)d_in[1];
    const float* bq = (const float*)d_in[2];
    const float* Wk = (const float*)d_in[3];
    const float* bk = (const float*)d_in[4];
    const float* Wv = (const float*)d_in[5];
    const float* bv = (const float*)d_in[6];
    const float* Wo = (const float*)d_in[7];
    const float* bo = (const float*)d_in[8];

    const int  M = 16384, N = 4096, K = 4096;
    const size_t MK = (size_t)M * K;
    const size_t NK = (size_t)N * K;

    bf16_t* xb  = (bf16_t*)d_ws;   // x in bf16
    bf16_t* wqb = xb  + MK;
    bf16_t* wkb = wqb + NK;
    bf16_t* wvb = wkb + NK;
    bf16_t* wob = wvb + NK;
    bf16_t* qb  = wob + NK;        // Q  [token][h*128+d]
    bf16_t* kb  = qb  + MK;        // K  [token][h*128+d]
    bf16_t* vtb = kb  + MK;        // V  [token][d*32+h] (token-transposed)
    bf16_t* aob = vtb + MK;        // attn output, bf16

    cvt_f32_bf16_kernel<<<(unsigned)((MK + 255) / 256), 256, 0, stream>>>(x,  xb,  (long long)MK);
    cvt_f32_bf16_kernel<<<(unsigned)((NK + 255) / 256), 256, 0, stream>>>(Wq, wqb, (long long)NK);
    cvt_f32_bf16_kernel<<<(unsigned)((NK + 255) / 256), 256, 0, stream>>>(Wk, wkb, (long long)NK);
    cvt_f32_bf16_kernel<<<(unsigned)((NK + 255) / 256), 256, 0, stream>>>(Wv, wvb, (long long)NK);
    cvt_f32_bf16_kernel<<<(unsigned)((NK + 255) / 256), 256, 0, stream>>>(Wo, wob, (long long)NK);

    dim3 ggrid(N / 128, M / 128);  // (32, 128)
    gemm_nt_wmma<0><<<ggrid, 256, 0, stream>>>(xb, wqb, bq, (void*)qb,  M, N, K);
    gemm_nt_wmma<0><<<ggrid, 256, 0, stream>>>(xb, wkb, bk, (void*)kb,  M, N, K);
    gemm_nt_wmma<1><<<ggrid, 256, 0, stream>>>(xb, wvb, bv, (void*)vtb, M, N, K);

    attention_wmma_kernel<<<M / 4, 128, 0, stream>>>(qb, kb, vtb, aob);

    gemm_nt_wmma<2><<<ggrid, 256, 0, stream>>>(aob, wob, bo, d_out, M, N, K);
}
